// KernelProjectionI2T_55267639165042
// MI455X (gfx1250) — compile-verified, matrix operand order checked by reading, not run on hardware
//
#include <hip/hip_runtime.h>
#include <math.h>

// ---------- types ----------
typedef __bf16 v16bf __attribute__((ext_vector_type(16)));
typedef float  v8f   __attribute__((ext_vector_type(8)));
typedef unsigned int u32x4 __attribute__((ext_vector_type(4)));

union BF16x16 { u32x4 q[2]; v16bf v; };

#define D_LAT 1024
#define RED   256
#define HW    36
#define NPAD  48            // 3 N-tiles of 16
#define TXT_STRIDE 1032     // 1024 + 8 pad (bank-conflict-free b128 rows)
#define EPSV 1e-8f

__device__ __forceinline__ unsigned short f2bf(float f) {
  union { float f; unsigned u; } c; c.f = f;
  unsigned u = c.u;
  unsigned r = u + 0x7FFFu + ((u >> 16) & 1u);   // round-to-nearest-even
  return (unsigned short)(r >> 16);
}

// ---------------------------------------------------------------------------
// K1: per-image i: qv = mean_hw(img), img_n = l2norm(qv), img_q = qv@red_w.T+red_b
// ---------------------------------------------------------------------------
__global__ __launch_bounds__(256) void k1_img(
    const float* __restrict__ img_embed, const float* __restrict__ red_w,
    const float* __restrict__ red_b, float* __restrict__ img_n,
    float* __restrict__ img_q) {
  __shared__ float sq[D_LAT];
  __shared__ float sred[256];
  const int i = blockIdx.x, tid = threadIdx.x;
  const float* base = img_embed + (size_t)i * HW * D_LAT;
  float ss = 0.f;
#pragma unroll
  for (int c = 0; c < 4; c++) {
    int d = tid + c * 256;
    float a = 0.f;
    for (int hw = 0; hw < HW; hw++) a += base[(size_t)hw * D_LAT + d];
    a *= (1.0f / 36.0f);
    sq[d] = a;
    ss += a * a;
  }
  sred[tid] = ss;
  __syncthreads();
  for (int off = 128; off > 0; off >>= 1) {
    if (tid < off) sred[tid] += sred[tid + off];
    __syncthreads();
  }
  float inv = 1.0f / (sqrtf(sred[0]) + EPSV);
#pragma unroll
  for (int c = 0; c < 4; c++) {
    int d = tid + c * 256;
    img_n[(size_t)i * D_LAT + d] = sq[d] * inv;
  }
  float acc = red_b[tid];
  const float* wr = red_w + (size_t)tid * D_LAT;
  for (int d = 0; d < D_LAT; d++) acc += sq[d] * wr[d];
  img_q[(size_t)i * RED + tid] = acc;
}

// ---------------------------------------------------------------------------
// K2a: w_lin[i, j] = img_q[i,:] @ proj_w[j,:] + proj_b[j]   (64 x 24576)
// ---------------------------------------------------------------------------
__global__ __launch_bounds__(256) void k2a_gen(
    const float* __restrict__ img_q, const float* __restrict__ proj_w,
    const float* __restrict__ proj_b, float* __restrict__ w_lin) {
  __shared__ float sq[16 * RED];
  const int j = blockIdx.x * 256 + threadIdx.x;
  const float pb = proj_b[j];
  const float* pw = proj_w + (size_t)j * RED;
#pragma unroll 1
  for (int it = 0; it < 4; it++) {
    __syncthreads();
    for (int e = threadIdx.x; e < 16 * RED; e += 256)
      sq[e] = img_q[(size_t)it * 16 * RED + e];
    __syncthreads();
    float acc[16];
#pragma unroll
    for (int ii = 0; ii < 16; ii++) acc[ii] = pb;
    for (int r = 0; r < RED; r++) {
      float w = pw[r];
#pragma unroll
      for (int ii = 0; ii < 16; ii++) acc[ii] += sq[ii * RED + r] * w;
    }
#pragma unroll
    for (int ii = 0; ii < 16; ii++)
      w_lin[(size_t)(it * 16 + ii) * 24576 + j] = acc[ii];
  }
}

// K2b: softmax over K=3 taps (in place)
__global__ __launch_bounds__(256) void k2b_softmax(float* __restrict__ w) {
  size_t idx = ((size_t)blockIdx.x * 256 + threadIdx.x) * 3;
  float a = w[idx], b = w[idx + 1], c = w[idx + 2];
  float m = fmaxf(a, fmaxf(b, c));
  float ea = __expf(a - m), eb = __expf(b - m), ec = __expf(c - m);
  float inv = 1.0f / (ea + eb + ec);
  w[idx] = ea * inv; w[idx + 1] = eb * inv; w[idx + 2] = ec * inv;
}

// ---------------------------------------------------------------------------
// K3: swizzle conv_w into bf16 WMMA A-fragment order
// ---------------------------------------------------------------------------
__global__ __launch_bounds__(256) void k3_swz(
    const float* __restrict__ conv_w, unsigned short* __restrict__ aswz) {
  int g = blockIdx.x * 256 + threadIdx.x;       // 2048 frags * 32 lanes
  int lane = g & 31, frag = g >> 5;
  int ks = frag & 31, mt = frag >> 5;
  int row = mt * 16 + (lane & 15);
  int kbase = ks * 32 + ((lane & 16) ? 8 : 0);
  const float* src = conv_w + (size_t)row * D_LAT;
  unsigned short* dst = aswz + ((size_t)frag * 32 + lane) * 16;
#pragma unroll
  for (int t = 0; t < 16; t++) {
    int kg = kbase + t + ((t & 8) ? 8 : 0);
    dst[t] = f2bf(src[kg]);
  }
}

// ---------------------------------------------------------------------------
// per-M-tile epilogue: masked max over t, bias, partial ssum/sdot
// C layout: lane<16 -> (M=v, N=lane); lane>=16 -> (M=v+8, N=lane-16)
// ---------------------------------------------------------------------------
__device__ __forceinline__ void reduce_tile(
    v8f t0, v8f t1, v8f t2, int mt, int nlo, bool hi,
    const float* __restrict__ conv_b, const float* __restrict__ imn,
    float& ssum, float& sdot) {
#pragma unroll
  for (int v = 0; v < 8; v++) {
    float mx = fmaxf(t0[v], t1[v]);                 // t = nlo, 16+nlo < 36
    mx = ((32 + nlo) < HW) ? fmaxf(mx, t2[v]) : mx; // t = 32+nlo masked
#pragma unroll
    for (int d = 1; d < 16; d <<= 1) mx = fmaxf(mx, __shfl_xor(mx, d, 32));
    int o2 = mt * 16 + v + (hi ? 8 : 0);
    float tv = mx + conv_b[o2];
    float in = imn[o2];
    bool active = (nlo == v);        // one lane per row per 16-lane half
    ssum += active ? tv * tv : 0.f;
    sdot += active ? in * tv : 0.f;
  }
}

// ---------------------------------------------------------------------------
// K4: fused per-(image,caption): dyn conv -> LDS(bf16) -> WMMA GEMM -> max -> sim
// ---------------------------------------------------------------------------
__global__ __launch_bounds__(256) void k4_main(
    const float* __restrict__ cap_embed, const float* __restrict__ w_dyn,
    const unsigned short* __restrict__ aswz, const float* __restrict__ conv_b,
    const float* __restrict__ img_n, float* __restrict__ out) {
  extern __shared__ char smem[];
  unsigned short* txt = (unsigned short*)smem;                // 48*1032 bf16
  float* capst = (float*)(smem + NPAD * TXT_STRIDE * 2);      // 8 waves * 288 f32
  float* accsh = capst + 8 * 288;                             // 2 f32

  const int i = blockIdx.x, b = blockIdx.y;
  const int tid = threadIdx.x, wave = tid >> 5, lane = tid & 31;

  if (tid < 2) accsh[tid] = 0.f;
  for (int e = tid; e < (NPAD - HW) * TXT_STRIDE; e += 256)
    txt[HW * TXT_STRIDE + e] = 0;

  // ---------------- phase 1: dynamic grouped conv -------------------------
  const float* capb = cap_embed + (size_t)b * 77 * D_LAT;
  const int o = lane >> 2, q = lane & 3;
  float* mycap = capst + wave * 288;
#pragma unroll 1
  for (int it = 0; it < 16; it++) {
    int gidx = wave * 16 + it;
    __syncthreads();
#pragma unroll
    for (int j = 0; j < 9; j++) {
      int e = lane + j * 32;
      int tt = e >> 3, din = e & 7;
      mycap[din * HW + tt] = capb[(size_t)tt * D_LAT + gidx * 8 + din];
    }
    float wk[24];
    const float* wp = w_dyn + (size_t)i * 24576 + (size_t)(gidx * 8 + o) * 24;
#pragma unroll
    for (int j = 0; j < 24; j++) wk[j] = wp[j];
    __syncthreads();
#pragma unroll
    for (int tq = 0; tq < 9; tq++) {
      int t = q * 9 + tq;
      float acc = 0.f;
#pragma unroll
      for (int din = 0; din < 8; din++) {
        float c0 = (t >= 1)     ? mycap[din * HW + t - 1] : 0.f;
        float c1 =                mycap[din * HW + t];
        float c2 = (t < HW - 1) ? mycap[din * HW + t + 1] : 0.f;
        acc += wk[din * 3 + 0] * c0 + wk[din * 3 + 1] * c1 + wk[din * 3 + 2] * c2;
      }
      txt[t * TXT_STRIDE + gidx * 8 + o] = f2bf(acc);
    }
  }
  __syncthreads();

  // ---------------- phase 2: WMMA GEMM + masked max + reduction -----------
  const int nlo = lane & 15;
  const bool hi = lane >= 16;
  const int dbaseOff = hi ? 8 : 0;
  float ssum = 0.f, sdot = 0.f;
  const float* imn = img_n + (size_t)i * D_LAT;
  // invariant B row bases (LDS): inner loop adds ks*32 elements (64B imm)
  const unsigned short* brow0 = txt + (0 * 16 + nlo) * TXT_STRIDE + dbaseOff;
  const unsigned short* brow1 = txt + (1 * 16 + nlo) * TXT_STRIDE + dbaseOff;
  const unsigned short* brow2 = txt + (2 * 16 + nlo) * TXT_STRIDE + dbaseOff;

#pragma unroll 1
  for (int p = 0; p < 4; p++) {
    const int mt0 = wave * 8 + p * 2;           // two consecutive M tiles
    // invariant A bases: fragment (mt,ks) lane slice at ((mt*32+ks)*32+lane)*16
    const unsigned short* a0 = aswz + ((size_t)mt0 * 1024 + lane) * 16;
    const unsigned short* a1 = a0 + 32 * 512;   // next M tile (32 frags * 512)
    v8f c00 = {}, c01 = {}, c02 = {}, c10 = {}, c11 = {}, c12 = {};
#pragma unroll 2
    for (int ks = 0; ks < 32; ks++) {
      BF16x16 A0, A1, B0, B1, B2;
      const unsigned short* ak0 = a0 + ks * 512;
      const unsigned short* ak1 = a1 + ks * 512;
      A0.q[0] = *(const u32x4*)(ak0);
      A0.q[1] = *(const u32x4*)(ak0 + 8);
      A1.q[0] = *(const u32x4*)(ak1);
      A1.q[1] = *(const u32x4*)(ak1 + 8);
      const int kd = ks * 32;
      B0.q[0] = *(const u32x4*)(brow0 + kd);
      B0.q[1] = *(const u32x4*)(brow0 + kd + 16);
      B1.q[0] = *(const u32x4*)(brow1 + kd);
      B1.q[1] = *(const u32x4*)(brow1 + kd + 16);
      B2.q[0] = *(const u32x4*)(brow2 + kd);
      B2.q[1] = *(const u32x4*)(brow2 + kd + 16);
      c00 = __builtin_amdgcn_wmma_f32_16x16x32_bf16(false, A0.v, false, B0.v, (short)0, c00, false, false);
      c01 = __builtin_amdgcn_wmma_f32_16x16x32_bf16(false, A0.v, false, B1.v, (short)0, c01, false, false);
      c02 = __builtin_amdgcn_wmma_f32_16x16x32_bf16(false, A0.v, false, B2.v, (short)0, c02, false, false);
      c10 = __builtin_amdgcn_wmma_f32_16x16x32_bf16(false, A1.v, false, B0.v, (short)0, c10, false, false);
      c11 = __builtin_amdgcn_wmma_f32_16x16x32_bf16(false, A1.v, false, B1.v, (short)0, c11, false, false);
      c12 = __builtin_amdgcn_wmma_f32_16x16x32_bf16(false, A1.v, false, B2.v, (short)0, c12, false, false);
    }
    reduce_tile(c00, c01, c02, mt0,     nlo, hi, conv_b, imn, ssum, sdot);
    reduce_tile(c10, c11, c12, mt0 + 1, nlo, hi, conv_b, imn, ssum, sdot);
  }
  // wave-wide sum, then block combine
#pragma unroll
  for (int d = 1; d < 32; d <<= 1) {
    ssum += __shfl_xor(ssum, d, 32);
    sdot += __shfl_xor(sdot, d, 32);
  }
  if (lane == 0) {
    atomicAdd(&accsh[0], ssum);
    atomicAdd(&accsh[1], sdot);
  }
  __syncthreads();
  if (tid == 0)
    out[(size_t)i * 64 + b] = accsh[1] / (sqrtf(accsh[0]) + EPSV);
}

// ---------------------------------------------------------------------------
extern "C" void kernel_launch(void* const* d_in, const int* in_sizes, int n_in,
                              void* d_out, int out_size, void* d_ws, size_t ws_size,
                              hipStream_t stream) {
  (void)in_sizes; (void)n_in; (void)out_size; (void)ws_size;
  const float* img_embed = (const float*)d_in[0];   // (64,36,1024)
  const float* cap_embed = (const float*)d_in[1];   // (64,77,1024)
  /* lens d_in[2] unused by the reference */
  const float* red_w  = (const float*)d_in[3];      // (256,1024)
  const float* red_b  = (const float*)d_in[4];      // (256)
  const float* proj_w = (const float*)d_in[5];      // (24576,256)
  const float* proj_b = (const float*)d_in[6];      // (24576)
  const float* conv_w = (const float*)d_in[7];      // (1024,1024)
  const float* conv_b = (const float*)d_in[8];      // (1024)
  float* out = (float*)d_out;                       // (64,64)

  float* img_n = (float*)d_ws;                          // 64*1024 f32
  float* img_q = img_n + 64 * D_LAT;                    // 64*256  f32
  float* w_dyn = img_q + 64 * RED;                      // 64*24576 f32
  unsigned short* aswz = (unsigned short*)(w_dyn + 64 * 24576); // 2048*512 bf16

  k1_img<<<64, 256, 0, stream>>>(img_embed, red_w, red_b, img_n, img_q);
  k2a_gen<<<96, 256, 0, stream>>>(img_q, proj_w, proj_b, w_dyn);
  k2b_softmax<<<2048, 256, 0, stream>>>(w_dyn);
  k3_swz<<<256, 256, 0, stream>>>(conv_w, aswz);

  size_t lds = (size_t)NPAD * TXT_STRIDE * 2 + 8 * 288 * 4 + 2 * 4;
  k4_main<<<dim3(64, 64), 256, lds, stream>>>(cap_embed, w_dyn, aswz, conv_b,
                                              img_n, out);
}
